// TransformerMultiHeadAttention_11166914969638
// MI455X (gfx1250) — matrix operand order repaired and checked by reading, not compiled, not run on hardware
//
#include <hip/hip_runtime.h>
#include <hip/hip_bf16.h>

// ---------------------------------------------------------------------------
// CDNA5 (gfx1250) multi-head causal attention
//   wave32 + v_wmma_f32_16x16x32_f16 + global_load_async_to_lds_b128
//   + DPP row reductions (no LDS crossbar in softmax)
// ---------------------------------------------------------------------------

typedef __attribute__((ext_vector_type(8)))  _Float16 v8h;
typedef __attribute__((ext_vector_type(16))) _Float16 v16h;
typedef __attribute__((ext_vector_type(4)))  float    v4f;
typedef __attribute__((ext_vector_type(8)))  float    v8f;

union Frag {
    v16h v;
    v8h  h[2];
};

static __device__ __forceinline__ v8f wmma_f16(const Frag& a, const Frag& b, v8f c) {
    return __builtin_amdgcn_wmma_f32_16x16x32_f16(
        /*neg_a=*/false, a.v, /*neg_b=*/false, b.v,
        /*c_mod=*/(short)0, c, /*reuse_a=*/false, /*reuse_b=*/false);
}

// CDNA5 async memory->LDS copy (16 bytes). Flat pointers into __shared__ carry
// the LDS byte offset in their low 32 bits (ISA 10.2 aperture mapping), which
// is the VDST operand of GLOBAL_LOAD_ASYNC_TO_LDS (tracked by ASYNCcnt).
static __device__ __forceinline__ void async_ld_b128(void* lds, const void* gptr) {
    uint32_t laddr = (uint32_t)(uintptr_t)lds;
    asm volatile("global_load_async_to_lds_b128 %0, %1, off"
                 :: "v"(laddr), "v"(gptr)
                 : "memory");
}
static __device__ __forceinline__ void async_fence() {
    asm volatile("s_wait_asynccnt 0x0" ::: "memory");
}

// DPP row_ror move within 16-lane rows (CTRL: 0x120 | n).
template <int CTRL>
static __device__ __forceinline__ float dpp_mov(float x) {
    int xi = __builtin_bit_cast(int, x);
    int r  = __builtin_amdgcn_update_dpp(xi, xi, CTRL, 0xF, 0xF, true);
    return __builtin_bit_cast(float, r);
}
// Max across a 16-lane row, VALU-only (rotate reduction).
static __device__ __forceinline__ float rowmax16(float x) {
    x = fmaxf(x, dpp_mov<0x128>(x));   // row_ror:8
    x = fmaxf(x, dpp_mov<0x124>(x));   // row_ror:4
    x = fmaxf(x, dpp_mov<0x122>(x));   // row_ror:2
    x = fmaxf(x, dpp_mov<0x121>(x));   // row_ror:1
    return x;
}

// ---------------------------------------------------------------------------
// Elementwise f32 -> f16 (activations): 8 elems/thread, b128 in, b128 out.
// ---------------------------------------------------------------------------
__global__ __launch_bounds__(256) void xconv(const float* __restrict__ X,
                                             _Float16* __restrict__ Y)
{
    size_t i = ((size_t)blockIdx.x * 256 + threadIdx.x) * 8;
    v4f a = *(const v4f*)(X + i);
    v4f b = *(const v4f*)(X + i + 4);
    v8h h;
#pragma unroll
    for (int e = 0; e < 4; ++e) { h[e] = (_Float16)a[e]; h[4 + e] = (_Float16)b[e]; }
    *(v8h*)(Y + i) = h;
}

// ---------------------------------------------------------------------------
// One-shot weight convert+transpose: f32 W -> f16 WT[N][K] (k contiguous).
//   HEADED: W stored [H, K, HD] (Wq/Wk/Wv); else plain [K, N] (Wo).
// ---------------------------------------------------------------------------
template <int HEADED>
__global__ __launch_bounds__(256) void wconv(const float* __restrict__ W,
                                             _Float16* __restrict__ WT)
{
    __shared__ alignas(16) _Float16 T[64][72];
    const int tid = threadIdx.x;
    const int k0 = blockIdx.y * 64, n0 = blockIdx.x * 64;

#pragma unroll
    for (int i = 0; i < 16; ++i) {
        int idx = tid + i * 256;
        int r = idx >> 6, c = idx & 63;          // r: k_local, c: n_local
        int k = k0 + r, n = n0 + c;
        size_t wi = HEADED ? ((size_t)((n >> 6) * 1024 + k) * 64 + (n & 63))
                           : ((size_t)k * 1024 + n);
        T[c][r] = (_Float16)W[wi];
    }
    __syncthreads();
#pragma unroll
    for (int i = 0; i < 2; ++i) {
        int seg = tid + i * 256;
        int r = seg >> 3, s8 = (seg & 7) * 8;
        *(v8h*)&WT[(size_t)(n0 + r) * 1024 + k0 + s8] = *(const v8h*)&T[r][s8];
    }
}

// ---------------------------------------------------------------------------
// WMMA GEMM:  out[M,N] = (A[M,K] * WT^T + bias) * oscale   (A, WT f16)
//   OUTMODE: 0 = f32 row-major [M,N] (d_out)
//            1 = f16 [B,H,S,HD]   (Q/K for attention; Q carries softmax scale)
//            2 = f16 [B,H,HD,S]   (V transposed for attention)
// Block 256 (8 waves), tile 128x64, K-chunk 64 (two 32-wide WMMA k-steps per
// LDS refill, halving barrier/fence rounds). All fills are async b128 copies.
// ---------------------------------------------------------------------------
template <int OUTMODE>
__global__ __launch_bounds__(256) void gemm_wmma(
    const _Float16* __restrict__ A, const _Float16* __restrict__ WT,
    const float* __restrict__ bias, void* __restrict__ outp, float oscale)
{
    constexpr int K = 1024;
    constexpr int BM = 128, BN = 64, BK = 64;
    __shared__ alignas(16) _Float16 As[BM][BK + 8];  // [m][k], 144B pitch
    __shared__ alignas(16) _Float16 Ws[BN][BK + 8];  // [n][k], 144B pitch

    const int tid    = threadIdx.x;
    const int lane   = tid & 31;
    const int wave   = tid >> 5;
    const int half16 = (lane >> 4) & 1;
    const int wm     = (wave >> 1) * 32;
    const int wn     = (wave & 1) * 32;
    const int bm     = blockIdx.y * BM;
    const int bn     = blockIdx.x * BN;

    v8f acc[2][2] = {};

    for (int k0 = 0; k0 < K; k0 += BK) {
        // --- W tile: 64x64 = 512 b128 segs (2/thread) ---
#pragma unroll
        for (int i = 0; i < 2; ++i) {
            int seg = tid + i * 256;
            int row = seg >> 3, sg = (seg & 7) * 8;
            async_ld_b128(&Ws[row][sg], WT + (size_t)(bn + row) * K + k0 + sg);
            if (k0 + BK < K)
                __builtin_prefetch(WT + (size_t)(bn + row) * K + k0 + BK + sg);
        }
        // --- A tile: 128x64 = 1024 b128 segs (4/thread) ---
#pragma unroll
        for (int i = 0; i < 4; ++i) {
            int seg = tid + i * 256;
            int row = seg >> 3, sg = (seg & 7) * 8;
            async_ld_b128(&As[row][sg], A + (size_t)(bm + row) * K + k0 + sg);
        }
        async_fence();
        __syncthreads();

#pragma unroll
        for (int kk = 0; kk < BK; kk += 32) {
            Frag afr[2], bfr[2];
#pragma unroll
            for (int t = 0; t < 2; ++t) {
                int row = wm + t * 16 + (lane & 15);
                int o = kk + half16 * 8;
                afr[t].h[0] = *(const v8h*)&As[row][o];
                afr[t].h[1] = *(const v8h*)&As[row][16 + o];
            }
#pragma unroll
            for (int t = 0; t < 2; ++t) {
                int nrow = wn + t * 16 + (lane & 15);
                int o = kk + half16 * 8;
                bfr[t].h[0] = *(const v8h*)&Ws[nrow][o];
                bfr[t].h[1] = *(const v8h*)&Ws[nrow][o + 16];
            }
#pragma unroll
            for (int i = 0; i < 2; ++i)
#pragma unroll
                for (int j = 0; j < 2; ++j)
                    acc[i][j] = wmma_f16(afr[i], bfr[j], acc[i][j]);
        }
        __syncthreads();
    }

#pragma unroll
    for (int i = 0; i < 2; ++i) {
#pragma unroll
        for (int j = 0; j < 2; ++j) {
            int n = bn + wn + j * 16 + (lane & 15);
            float bv = bias[n];
#pragma unroll
            for (int r = 0; r < 8; ++r) {
                int m = bm + wm + i * 16 + r + half16 * 8;
                float val = (acc[i][j][r] + bv) * oscale;
                if (OUTMODE == 0) {
                    ((float*)outp)[(size_t)m * 1024 + n] = val;
                } else {
                    int b = m >> 11, s = m & 2047;
                    int h = n >> 6,  d = n & 63;
                    if (OUTMODE == 1)
                        ((_Float16*)outp)[((size_t)(b * 16 + h) * 2048 + s) * 64 + d] =
                            (_Float16)val;
                    else  // V^T: [B,H,HD,S]
                        ((_Float16*)outp)[((size_t)((b * 16 + h) * 64 + d)) * 2048 + s] =
                            (_Float16)val;
                }
            }
        }
    }
}

// ---------------------------------------------------------------------------
// Causal flash attention. Q,K f16 [B,H,S,HD] (Q pre-scaled by log2e/sqrt(HD));
// V f16 [B,H,HD,S]. Grid (S/64, B*H); block 128 = 4 waves; wave owns 16 q rows.
// Softmax in exp2 domain; row max via DPP rotate reduction; row sums via an
// extra WMMA tile against a constant ones matrix.
// ---------------------------------------------------------------------------
__global__ __launch_bounds__(128) void flash_attn(
    const _Float16* __restrict__ Q, const _Float16* __restrict__ Kmat,
    const _Float16* __restrict__ Vtg, _Float16* __restrict__ O)
{
    constexpr int S = 2048, HD = 64;
    const int bh     = blockIdx.y;
    const int qb     = blockIdx.x * 64;
    const int tid    = threadIdx.x;
    const int lane   = tid & 31;
    const int wave   = tid >> 5;
    const int half16 = (lane >> 4) & 1;
    const int qrow   = qb + wave * 16;

    __shared__ alignas(16) _Float16 Ks[32][72];     // [s_local][d], 144B pitch
    __shared__ alignas(16) _Float16 Vt[64][40];     // [d][s_local], 80B pitch
    __shared__ alignas(16) _Float16 Ps[4][16][40];  // per-wave P staging

    const _Float16* Qb = Q    + (size_t)bh * S * HD;
    const _Float16* Kb = Kmat + (size_t)bh * S * HD;
    const _Float16* Vb = Vtg  + (size_t)bh * HD * S;

    Frag qf[2];
    {
        const _Float16* qr = Qb + (size_t)(qrow + (lane & 15)) * HD;
#pragma unroll
        for (int kd = 0; kd < 2; ++kd) {
            int o = kd * 32 + half16 * 8;
            qf[kd].h[0] = *(const v8h*)(qr + o);
            qf[kd].h[1] = *(const v8h*)(qr + o + 16);
        }
    }

    Frag fones;                                     // B = ones(32x16): row sums
#pragma unroll
    for (int e = 0; e < 16; ++e) fones.v[e] = (_Float16)1.0f;

    v8f  o_acc[5] = {};                             // [0..3] output, [4] = l
    float m_run[8];
#pragma unroll
    for (int r = 0; r < 8; ++r) m_run[r] = -3.0e38f;

    auto process = [&](int s0, bool domask) {
        // --- K tile: 256 b128 segs (2/thread); V^T tile: 256 segs (2/thread) ---
#pragma unroll
        for (int i = 0; i < 2; ++i) {
            int seg = tid + i * 128;
            int r = seg >> 3, sg = (seg & 7) * 8;
            async_ld_b128(&Ks[r][sg], Kb + (size_t)(s0 + r) * HD + sg);
        }
#pragma unroll
        for (int i = 0; i < 2; ++i) {
            int seg = tid + i * 128;
            int d = seg >> 2, sg = (seg & 3) * 8;
            async_ld_b128(&Vt[d][sg], Vb + (size_t)d * S + s0 + sg);
        }
        async_fence();
        __syncthreads();

        // --- scores (already in exp2 domain; Q carries the scale) ---
        v8f sc[2] = {};
#pragma unroll
        for (int t = 0; t < 2; ++t) {
#pragma unroll
            for (int kd = 0; kd < 2; ++kd) {
                Frag bf;
                int krow = t * 16 + (lane & 15);
                int o = kd * 32 + half16 * 8;
                bf.h[0] = *(const v8h*)&Ks[krow][o];
                bf.h[1] = *(const v8h*)&Ks[krow][o + 16];
                sc[t] = wmma_f16(qf[kd], bf, sc[t]);
            }
        }

        // --- (diagonal only) causal mask; row max via DPP rotate reduction ---
        float mnew[8];
#pragma unroll
        for (int r = 0; r < 8; ++r) {
            if (domask) {
                int qg = qrow + r + half16 * 8;
#pragma unroll
                for (int t = 0; t < 2; ++t) {
                    int sg = s0 + t * 16 + (lane & 15);
                    if (sg > qg) sc[t][r] = -3.0e38f;
                }
            }
            mnew[r] = rowmax16(fmaxf(sc[0][r], sc[1][r]));
        }

        // --- online softmax update (exp2 domain); l via ones-WMMA tile ---
#pragma unroll
        for (int r = 0; r < 8; ++r) {
            float mN   = fmaxf(m_run[r], mnew[r]);
            float corr = exp2f(m_run[r] - mN);
            m_run[r] = mN;
#pragma unroll
            for (int t = 0; t < 2; ++t)
                sc[t][r] = exp2f(sc[t][r] - mN);
#pragma unroll
            for (int tn = 0; tn < 5; ++tn)
                o_acc[tn][r] *= corr;
        }

        // --- reshape P (C/D layout) -> A layout via per-wave LDS staging ---
#pragma unroll
        for (int r = 0; r < 8; ++r) {
            int row = r + half16 * 8;
            Ps[wave][row][(lane & 15)]      = (_Float16)sc[0][r];
            Ps[wave][row][16 + (lane & 15)] = (_Float16)sc[1][r];
        }
        Frag pf;
        {
            int row = lane & 15;
            int o = half16 * 8;
            pf.h[0] = *(const v8h*)&Ps[wave][row][o];
            pf.h[1] = *(const v8h*)&Ps[wave][row][16 + o];
        }

        // --- PV (4 tiles) + row-sum tile ---
#pragma unroll
        for (int tn = 0; tn < 4; ++tn) {
            Frag bf;
            int drow = tn * 16 + (lane & 15);
            int o = half16 * 8;
            bf.h[0] = *(const v8h*)&Vt[drow][o];
            bf.h[1] = *(const v8h*)&Vt[drow][o + 16];
            o_acc[tn] = wmma_f16(pf, bf, o_acc[tn]);
        }
        o_acc[4] = wmma_f16(pf, fones, o_acc[4]);
        __syncthreads();
    };

    const int nfull = qb / 32;                 // blocks unmasked for all waves
    for (int j = 0; j < nfull; ++j) process(j * 32, false);
    process(qb, true);                         // diagonal blocks
    process(qb + 32, true);

    // --- normalize and store heads-concatenated f16 [B,S,D] ---
    const int b = bh >> 4, h = bh & 15;
#pragma unroll
    for (int r = 0; r < 8; ++r) {
        int qg = qrow + r + half16 * 8;
        float inv = 1.0f / o_acc[4][r];        // every lane holds its row's l
#pragma unroll
        for (int tn = 0; tn < 4; ++tn) {
            int d = tn * 16 + (lane & 15);
            O[((size_t)(b * S + qg)) * 1024 + h * 64 + d] =
                (_Float16)(o_acc[tn][r] * inv);
        }
    }
}

// ---------------------------------------------------------------------------
// Inputs: 0:q 1:k 2:v 3:mask 4:Wq 5:bq 6:Wk 7:bk 8:Wv 9:bv 10:Wo 11:bo
// ws (f16 halves): WTq|WTk|WTv|WTo (1M each) | Xq|Xk|Xv f16 (4M each)
//                  | Q|K|V (4M each) ; attn reuses the Xq slot.  56 MB total.
// ---------------------------------------------------------------------------
extern "C" void kernel_launch(void* const* d_in, const int* in_sizes, int n_in,
                              void* d_out, int out_size, void* d_ws, size_t ws_size,
                              hipStream_t stream) {
    (void)in_sizes; (void)n_in; (void)out_size; (void)ws_size;
    const float* Xq = (const float*)d_in[0];
    const float* Xk = (const float*)d_in[1];
    const float* Xv = (const float*)d_in[2];
    const float* Wq = (const float*)d_in[4];
    const float* bq = (const float*)d_in[5];
    const float* Wk = (const float*)d_in[6];
    const float* bk = (const float*)d_in[7];
    const float* Wv = (const float*)d_in[8];
    const float* bv = (const float*)d_in[9];
    const float* Wo = (const float*)d_in[10];
    const float* bo = (const float*)d_in[11];

    const size_t WSZ = (size_t)1024 * 1024;
    const size_t MAT = (size_t)4096 * 1024;
    _Float16* wt   = (_Float16*)d_ws;
    _Float16* wtq  = wt;
    _Float16* wtk  = wt + WSZ;
    _Float16* wtv  = wt + 2 * WSZ;
    _Float16* wto  = wt + 3 * WSZ;
    _Float16* xq16 = wt + 4 * WSZ;
    _Float16* xk16 = xq16 + MAT;
    _Float16* xv16 = xk16 + MAT;
    _Float16* wsQ  = xv16 + MAT;
    _Float16* wsK  = wsQ + MAT;
    _Float16* wsV  = wsK + MAT;
    _Float16* wsA  = xq16;            // reuse: Xq f16 dead after projections

    dim3 wg(16, 16);
    wconv<1><<<wg, 256, 0, stream>>>(Wq, wtq);
    wconv<1><<<wg, 256, 0, stream>>>(Wk, wtk);
    wconv<1><<<wg, 256, 0, stream>>>(Wv, wtv);
    wconv<0><<<wg, 256, 0, stream>>>(Wo, wto);

    const int xblocks = (int)(MAT / (256 * 8));   // 2048
    xconv<<<xblocks, 256, 0, stream>>>(Xq, xq16);
    xconv<<<xblocks, 256, 0, stream>>>(Xk, xk16);
    xconv<<<xblocks, 256, 0, stream>>>(Xv, xv16);

    const float qscale = 0.125f * 1.4426950408889634f;  // 1/sqrt(64) * log2(e)
    dim3 gg(16, 32);
    gemm_wmma<1><<<gg, 256, 0, stream>>>(xq16, wtq, bq, wsQ, qscale);
    gemm_wmma<1><<<gg, 256, 0, stream>>>(xk16, wtk, bk, wsK, 1.0f);
    gemm_wmma<2><<<gg, 256, 0, stream>>>(xv16, wtv, bv, wsV, 1.0f);

    dim3 fg(32, 32);
    flash_attn<<<fg, 128, 0, stream>>>(wsQ, wsK, wsV, wsA);

    gemm_wmma<0><<<gg, 256, 0, stream>>>(wsA, wto, bo, (float*)d_out, 1.0f);
}
